// GuidedUpsampleUnitGF_14362370638121
// MI455X (gfx1250) — matrix operand order repaired and checked by reading, not compiled, not run on hardware
//
#include <hip/hip_runtime.h>
#include <hip/hip_bf16.h>
#include <cstdint>
#include <cstddef>

// ---------------------------------------------------------------------------
// GuidedUpsampleUnitGF for MI455X (gfx1250, wave32, WMMA f16 16x16x32)
//
// Pipeline: bilinear 2x upsample+concat -> 1x1 conv (WMMA) -> 5x5 conv
// (implicit GEMM, WMMA) -> guided filter (r=2, box means w/ border counts)
// -> 1x1 conv (WMMA) -> mask head (dot+sigmoid).
// All matmuls: f32 -> f16 operands, f32 accumulation via
// __builtin_amdgcn_wmma_f32_16x16x32_f16. Activations stay L2-resident
// (192MB L2 >> 64MB biggest tensor), so fragments load direct from global.
// Wave tile: 64x64 (4 A-frags x 4 B-frags = 16 WMMA/k-step, ~33 FLOP/L2-byte).
// ---------------------------------------------------------------------------

typedef _Float16 half_t;
typedef __attribute__((ext_vector_type(8)))  _Float16 v8h;
typedef __attribute__((ext_vector_type(16))) _Float16 v16h;
typedef __attribute__((ext_vector_type(8)))  float    v8f;

union FragU { v16h f; struct { v8h lo, hi; } s; };

#define H2  256
#define W2  256
#define HW  65536       // 256*256 pixels
#define CH  256         // embed channels
#define K1P 288         // 263 input channels padded to 9*32
#define K5  6400        // 25 taps * 256 channels

__device__ __forceinline__ v8f wmma16(v16h a, v16h b, v8f c) {
  // D = A(16x32 f16) * B(32x16 f16) + C(16x16 f32)
  return __builtin_amdgcn_wmma_f32_16x16x32_f16(
      /*neg_a=*/false, a, /*neg_b=*/false, b,
      /*c_mod=*/(short)0, c, /*reuse_a=*/false, /*reuse_b=*/false);
}

// A-matrix fragment (16x32 f16): lane m=l%16 g=l/16; halves h -> A[m][16g+h]
// (contiguous 16 halves per lane, matching ldB). Row-major source.
__device__ __forceinline__ v16h ldA(const half_t* __restrict__ base, int stride) {
  const int l = threadIdx.x & 31;
  const half_t* p = base + (size_t)(l & 15) * stride + ((l >> 4) << 4);
  FragU u;
  u.s.lo = *(const v8h*)p;
  u.s.hi = *(const v8h*)(p + 8);
  return u.f;
}

// B-matrix fragment (32x16 f16): lane n=l%16 g=l/16; halves h -> K=h+16g
// (contiguous 16 halves per lane). Source stored [col n][K] row-major.
__device__ __forceinline__ v16h ldB(const half_t* __restrict__ base, int stride) {
  const int l = threadIdx.x & 31;
  const half_t* p = base + (size_t)(l & 15) * stride + ((l >> 4) << 4);
  FragU u;
  u.s.lo = *(const v8h*)p;
  u.s.hi = *(const v8h*)(p + 8);
  return u.f;
}

// Predicated B fragment: per-lane pointer already resolved; zero when invalid
// (halo / padding pixels of the 5x5 conv).
__device__ __forceinline__ v16h ldBpred(const half_t* __restrict__ p, bool ok) {
  FragU u;
  if (ok) {
    u.s.lo = *(const v8h*)p;
    u.s.hi = *(const v8h*)(p + 8);
  } else {
    v8h z = {};
    u.s.lo = z;
    u.s.hi = z;
  }
  return u.f;
}

// ---------------------------------------------------------------------------
// Weight prep: cast to f16 in GEMM-friendly layouts.
//   a1[256][288]  <- w1 (256 x 263), zero padded
//   a2[256][6400] <- w2 (256,256,5,5) with k = tap*256 + ci
//   a3[256][256]  <- w3
// ---------------------------------------------------------------------------
__global__ __launch_bounds__(256) void k_prepw(
    const float* __restrict__ w1, const float* __restrict__ w2,
    const float* __restrict__ w3, half_t* __restrict__ a1,
    half_t* __restrict__ a2, half_t* __restrict__ a3) {
  const int idx = blockIdx.x * 256 + threadIdx.x;
  if (idx < 256 * K5) {
    const int o = idx / K5, k = idx % K5;
    const int tap = k >> 8, ci = k & 255;
    a2[idx] = (half_t)w2[((size_t)o * 256 + ci) * 25 + tap];
  }
  if (idx < 256 * K1P) {
    const int o = idx / K1P, k = idx % K1P;
    a1[idx] = (half_t)((k < 263) ? w1[(size_t)o * 263 + k] : 0.0f);
  }
  if (idx < 256 * 256) a3[idx] = (half_t)w3[idx];
}

// ---------------------------------------------------------------------------
// Bilinear 2x upsample (half-pixel centers) of mask/fg/bg/feat fused into
// xbuf[pixel][288] (f16), plus guide = mean(fg_2x, ch) * 128.
// ---------------------------------------------------------------------------
__global__ __launch_bounds__(256) void k_prep(
    const float* __restrict__ fg, const float* __restrict__ bg,
    const float* __restrict__ mask, const float* __restrict__ feat,
    half_t* __restrict__ xbuf, float* __restrict__ guide) {
  const int pix = blockIdx.x * 256 + threadIdx.x;
  const int y = pix >> 8, x = pix & 255;
  const float fy = 0.5f * (float)y - 0.25f;
  const float fx = 0.5f * (float)x - 0.25f;
  const float y0f = floorf(fy), x0f = floorf(fx);
  const float wy = fy - y0f, wx = fx - x0f;
  const int iy0 = min(127, max(0, (int)y0f));
  const int iy1 = min(127, max(0, (int)y0f + 1));
  const int ix0 = min(127, max(0, (int)x0f));
  const int ix1 = min(127, max(0, (int)x0f + 1));
  const float w00 = (1.f - wy) * (1.f - wx), w01 = (1.f - wy) * wx;
  const float w10 = wy * (1.f - wx), w11 = wy * wx;
  const int i00 = iy0 * 128 + ix0, i01 = iy0 * 128 + ix1;
  const int i10 = iy1 * 128 + ix0, i11 = iy1 * 128 + ix1;

  auto bil = [&](const float* __restrict__ s) -> float {
    return w00 * s[i00] + w01 * s[i01] + w10 * s[i10] + w11 * s[i11];
  };

  half_t* xr = xbuf + (size_t)pix * K1P;
  xr[0] = (half_t)bil(mask);
  float gsum = 0.f;
  #pragma unroll
  for (int c = 0; c < 3; ++c) {
    const float v = bil(fg + c * 16384);
    xr[1 + c] = (half_t)v;
    gsum += v;
  }
  #pragma unroll
  for (int c = 0; c < 3; ++c) xr[4 + c] = (half_t)bil(bg + c * 16384);
  for (int c = 0; c < 256; ++c) xr[7 + c] = (half_t)bil(feat + c * 16384);
  #pragma unroll
  for (int c = 263; c < K1P; ++c) xr[c] = (half_t)0.f;

  guide[pix] = gsum * (128.0f / 3.0f);
}

// ---------------------------------------------------------------------------
// GEMM1: relu(w1 @ x + b1). M=256 ch, N=65536 pix, K=288.
// Block = 8 waves: 4(M) x 2(N); wave tile 64x64 = 4x4 WMMA tiles.
// Output y1 f16 [pixel][256] (native B layout for next conv).
// ---------------------------------------------------------------------------
__global__ __launch_bounds__(256) void k_gemm1(
    const half_t* __restrict__ A, const half_t* __restrict__ B,
    const float* __restrict__ bias, half_t* __restrict__ Y) {
  const int lane = threadIdx.x & 31;
  const int wave = threadIdx.x >> 5;
  const int wm = wave & 3, wn = wave >> 2;
  const int mbase = wm * 64;
  const int nbase = blockIdx.x * 128 + wn * 64;
  const int g = lane >> 4, nl = lane & 15;

  v8f zf = {};
  v8f acc[4][4];
  #pragma unroll
  for (int i = 0; i < 4; ++i)
    #pragma unroll
    for (int j = 0; j < 4; ++j) acc[i][j] = zf;

  for (int kc = 0; kc < 9; ++kc) {
    const int ko = kc * 32;
    v16h a[4], b[4];
    #pragma unroll
    for (int i = 0; i < 4; ++i)
      a[i] = ldA(A + (size_t)(mbase + i * 16) * K1P + ko, K1P);
    #pragma unroll
    for (int j = 0; j < 4; ++j)
      b[j] = ldB(B + (size_t)(nbase + j * 16) * K1P + ko, K1P);
    #pragma unroll
    for (int i = 0; i < 4; ++i)
      #pragma unroll
      for (int j = 0; j < 4; ++j) acc[i][j] = wmma16(a[i], b[j], acc[i][j]);
  }

  #pragma unroll
  for (int mt = 0; mt < 4; ++mt) {
    #pragma unroll
    for (int j = 0; j < 4; ++j) {
      const int pix = nbase + j * 16 + nl;
      const int ch0 = mbase + mt * 16 + g * 8;
      v8h o;
      #pragma unroll
      for (int v = 0; v < 8; ++v) {
        const float t = acc[mt][j][v] + bias[ch0 + v];
        o[v] = (half_t)(t > 0.f ? t : 0.f);
      }
      *(v8h*)(Y + (size_t)pix * CH + ch0) = o;
    }
  }
}

// ---------------------------------------------------------------------------
// GEMM5: relu(conv5x5(y1, w2) + b2), implicit GEMM. K = 25 taps x 256 ch.
// N-tile = 128 consecutive pixels of one image row -> uniform dy handling,
// per-lane dx halo predication, zero fill at borders.
// Output P f32 [channel][pixel] (for guided filter).
// ---------------------------------------------------------------------------
__global__ __launch_bounds__(256) void k_gemm5(
    const half_t* __restrict__ A, const half_t* __restrict__ Y1,
    const float* __restrict__ bias, float* __restrict__ P) {
  const int lane = threadIdx.x & 31;
  const int wave = threadIdx.x >> 5;
  const int wm = wave & 3, wn = wave >> 2;
  const int mbase = wm * 64;
  const int pb = blockIdx.x * 128;
  const int y = pb >> 8;
  const int xbw = (pb & 255) + wn * 64;
  const int g = lane >> 4, nl = lane & 15;

  v8f zf = {};
  v8f acc[4][4];
  #pragma unroll
  for (int i = 0; i < 4; ++i)
    #pragma unroll
    for (int j = 0; j < 4; ++j) acc[i][j] = zf;

  for (int tap = 0; tap < 25; ++tap) {
    const int dy = tap / 5 - 2, dx = tap % 5 - 2;
    const int yy = y + dy;
    const bool rowok = ((unsigned)yy < 256u);
    for (int kc = 0; kc < 8; ++kc) {
      const int ko = tap * 256 + kc * 32;
      v16h a[4], b[4];
      #pragma unroll
      for (int i = 0; i < 4; ++i)
        a[i] = ldA(A + (size_t)(mbase + i * 16) * K5 + ko, K5);
      #pragma unroll
      for (int j = 0; j < 4; ++j) {
        const int xx = xbw + j * 16 + nl + dx;
        const bool ok = rowok && ((unsigned)xx < 256u);
        const half_t* p =
            Y1 + (size_t)(yy * 256 + xx) * CH + kc * 32 + (g << 4);
        b[j] = ldBpred(p, ok);
      }
      #pragma unroll
      for (int i = 0; i < 4; ++i)
        #pragma unroll
        for (int j = 0; j < 4; ++j) acc[i][j] = wmma16(a[i], b[j], acc[i][j]);
    }
  }

  #pragma unroll
  for (int mt = 0; mt < 4; ++mt) {
    #pragma unroll
    for (int j = 0; j < 4; ++j) {
      const int pix = pb + wn * 64 + j * 16 + nl;
      const int ch0 = mbase + mt * 16 + g * 8;
      #pragma unroll
      for (int v = 0; v < 8; ++v) {
        const float t = acc[mt][j][v] + bias[ch0 + v];
        P[(size_t)(ch0 + v) * HW + pix] = (t > 0.f ? t : 0.f);
      }
    }
  }
}

// ---------------------------------------------------------------------------
// Guided filter stage 1: per-pixel mean(I) and 1/(var(I)+eps) over the
// clipped 5x5 window (normalized by valid count = matches cumsum reference).
// ---------------------------------------------------------------------------
__global__ __launch_bounds__(256) void k_gfstats(
    const float* __restrict__ guide, float* __restrict__ meanI,
    float* __restrict__ invv) {
  const int pix = blockIdx.x * 256 + threadIdx.x;
  const int y = pix >> 8, x = pix & 255;
  const int y0 = max(y - 2, 0), y1 = min(y + 2, 255);
  const int x0 = max(x - 2, 0), x1 = min(x + 2, 255);
  float sI = 0.f, sII = 0.f;
  for (int yy = y0; yy <= y1; ++yy)
    for (int xx = x0; xx <= x1; ++xx) {
      const float v = guide[yy * 256 + xx];
      sI += v;
      sII += v * v;
    }
  const float cnt = (float)((y1 - y0 + 1) * (x1 - x0 + 1));
  const float mi = sI / cnt;
  meanI[pix] = mi;
  invv[pix] = 1.0f / ((sII / cnt - mi * mi) + 0.01f);
}

// ---------------------------------------------------------------------------
// Guided filter stage 2: per channel z, LDS-tiled box means of p and I*p ->
// a, b coefficients (stored f16, channel-major).
// ---------------------------------------------------------------------------
__global__ __launch_bounds__(256) void k_gfab(
    const float* __restrict__ P, const float* __restrict__ guide,
    const float* __restrict__ meanI, const float* __restrict__ invv,
    half_t* __restrict__ aF, half_t* __restrict__ bF) {
  __shared__ float sI[20][20];
  __shared__ float sP[20][20];
  const int z = blockIdx.z;
  const int oy = blockIdx.y * 16 - 2, ox = blockIdx.x * 16 - 2;
  const int tid = threadIdx.y * 16 + threadIdx.x;
  const float* Pz = P + (size_t)z * HW;
  for (int i = tid; i < 400; i += 256) {
    const int ly = i / 20, lx = i % 20;
    const int yy = oy + ly, xx = ox + lx;
    const bool in = ((unsigned)yy < 256u) && ((unsigned)xx < 256u);
    sI[ly][lx] = in ? guide[yy * 256 + xx] : 0.f;
    sP[ly][lx] = in ? Pz[yy * 256 + xx] : 0.f;
  }
  __syncthreads();

  const int x = blockIdx.x * 16 + threadIdx.x;
  const int y = blockIdx.y * 16 + threadIdx.y;
  float sp = 0.f, sip = 0.f;
  #pragma unroll
  for (int dy = 0; dy < 5; ++dy)
    #pragma unroll
    for (int dx = 0; dx < 5; ++dx) {
      const float pv = sP[threadIdx.y + dy][threadIdx.x + dx];
      sp += pv;
      sip += sI[threadIdx.y + dy][threadIdx.x + dx] * pv;
    }
  const float cnt = (float)((min(y + 2, 255) - max(y - 2, 0) + 1) *
                            (min(x + 2, 255) - max(x - 2, 0) + 1));
  const int pix = y * 256 + x;
  const float mp = sp / cnt, mip = sip / cnt;
  const float mi = meanI[pix];
  const float a = (mip - mi * mp) * invv[pix];
  const float b = mp - a * mi;
  aF[(size_t)z * HW + pix] = (half_t)a;
  bF[(size_t)z * HW + pix] = (half_t)b;
}

// ---------------------------------------------------------------------------
// Guided filter stage 3: q = boxmean(a)*I + boxmean(b); store f16 in
// [pixel][channel] layout (native WMMA-B layout for the next 1x1 conv).
// ---------------------------------------------------------------------------
__global__ __launch_bounds__(256) void k_gfq(
    const half_t* __restrict__ aF, const half_t* __restrict__ bF,
    const float* __restrict__ guide, half_t* __restrict__ filt) {
  __shared__ float sA[20][20];
  __shared__ float sB[20][20];
  const int z = blockIdx.z;
  const int oy = blockIdx.y * 16 - 2, ox = blockIdx.x * 16 - 2;
  const int tid = threadIdx.y * 16 + threadIdx.x;
  const half_t* Az = aF + (size_t)z * HW;
  const half_t* Bz = bF + (size_t)z * HW;
  for (int i = tid; i < 400; i += 256) {
    const int ly = i / 20, lx = i % 20;
    const int yy = oy + ly, xx = ox + lx;
    const bool in = ((unsigned)yy < 256u) && ((unsigned)xx < 256u);
    sA[ly][lx] = in ? (float)Az[yy * 256 + xx] : 0.f;
    sB[ly][lx] = in ? (float)Bz[yy * 256 + xx] : 0.f;
  }
  __syncthreads();

  const int x = blockIdx.x * 16 + threadIdx.x;
  const int y = blockIdx.y * 16 + threadIdx.y;
  float sa = 0.f, sb = 0.f;
  #pragma unroll
  for (int dy = 0; dy < 5; ++dy)
    #pragma unroll
    for (int dx = 0; dx < 5; ++dx) {
      sa += sA[threadIdx.y + dy][threadIdx.x + dx];
      sb += sB[threadIdx.y + dy][threadIdx.x + dx];
    }
  const float cnt = (float)((min(y + 2, 255) - max(y - 2, 0) + 1) *
                            (min(x + 2, 255) - max(x - 2, 0) + 1));
  const int pix = y * 256 + x;
  const float q = (sa / cnt) * guide[pix] + sb / cnt;
  filt[(size_t)pix * CH + z] = (half_t)q;
}

// ---------------------------------------------------------------------------
// GEMM3: feat_2x = relu(w3 @ filt + b3), f32 output channel-major (final
// output layout (1,256,256,256)).
// ---------------------------------------------------------------------------
__global__ __launch_bounds__(256) void k_gemm3(
    const half_t* __restrict__ A, const half_t* __restrict__ B,
    const float* __restrict__ bias, float* __restrict__ OUT) {
  const int lane = threadIdx.x & 31;
  const int wave = threadIdx.x >> 5;
  const int wm = wave & 3, wn = wave >> 2;
  const int mbase = wm * 64;
  const int nbase = blockIdx.x * 128 + wn * 64;
  const int g = lane >> 4, nl = lane & 15;

  v8f zf = {};
  v8f acc[4][4];
  #pragma unroll
  for (int i = 0; i < 4; ++i)
    #pragma unroll
    for (int j = 0; j < 4; ++j) acc[i][j] = zf;

  for (int kc = 0; kc < 8; ++kc) {
    const int ko = kc * 32;
    v16h a[4], b[4];
    #pragma unroll
    for (int i = 0; i < 4; ++i)
      a[i] = ldA(A + (size_t)(mbase + i * 16) * CH + ko, CH);
    #pragma unroll
    for (int j = 0; j < 4; ++j)
      b[j] = ldB(B + (size_t)(nbase + j * 16) * CH + ko, CH);
    #pragma unroll
    for (int i = 0; i < 4; ++i)
      #pragma unroll
      for (int j = 0; j < 4; ++j) acc[i][j] = wmma16(a[i], b[j], acc[i][j]);
  }

  #pragma unroll
  for (int mt = 0; mt < 4; ++mt) {
    #pragma unroll
    for (int j = 0; j < 4; ++j) {
      const int pix = nbase + j * 16 + nl;
      const int ch0 = mbase + mt * 16 + g * 8;
      #pragma unroll
      for (int v = 0; v < 8; ++v) {
        const float t = acc[mt][j][v] + bias[ch0 + v];
        OUT[(size_t)(ch0 + v) * HW + pix] = (t > 0.f ? t : 0.f);
      }
    }
  }
}

// ---------------------------------------------------------------------------
// Mask head: sigmoid(wm . feat_2x + bm) per pixel.
// ---------------------------------------------------------------------------
__global__ __launch_bounds__(256) void k_mask(
    const float* __restrict__ feat, const float* __restrict__ wmw,
    const float* __restrict__ bm, float* __restrict__ outm) {
  const int pix = blockIdx.x * 256 + threadIdx.x;
  float s = bm[0];
  for (int c = 0; c < 256; ++c) s += wmw[c] * feat[(size_t)c * HW + pix];
  outm[pix] = 1.0f / (1.0f + __expf(-s));
}

// ---------------------------------------------------------------------------
// Workspace layout (bytes). Peak ~169 MiB with buffer reuse:
//   xbuf region reused for a-coeffs (xbuf dead after gemm1),
//   y1 region reused for filt (y1 dead after gemm5).
// ---------------------------------------------------------------------------
static constexpr size_t OFF_XBUF  = 0;                          // 37,748,736 (f16 [HW][288]); later aF16
static constexpr size_t OFF_A1    = OFF_XBUF  + (size_t)HW * K1P * 2;
static constexpr size_t OFF_A2    = OFF_A1    + (size_t)256 * K1P * 2;
static constexpr size_t OFF_A3    = OFF_A2    + (size_t)256 * K5 * 2;
static constexpr size_t OFF_Y1    = OFF_A3    + (size_t)256 * 256 * 2; // f16 [HW][256]; later filt
static constexpr size_t OFF_GUIDE = OFF_Y1    + (size_t)HW * CH * 2;
static constexpr size_t OFF_MEANI = OFF_GUIDE + (size_t)HW * 4;
static constexpr size_t OFF_INV   = OFF_MEANI + (size_t)HW * 4;
static constexpr size_t OFF_P     = OFF_INV   + (size_t)HW * 4;        // f32 [256][HW]
static constexpr size_t OFF_BF    = OFF_P     + (size_t)CH * HW * 4;   // f16 [256][HW]
// total = OFF_BF + HW*CH*2  ~= 176,308,224 bytes

extern "C" void kernel_launch(void* const* d_in, const int* in_sizes, int n_in,
                              void* d_out, int out_size, void* d_ws, size_t ws_size,
                              hipStream_t stream) {
  (void)in_sizes; (void)n_in; (void)out_size; (void)ws_size;
  const float* fg   = (const float*)d_in[0];
  const float* bg   = (const float*)d_in[1];
  const float* mask = (const float*)d_in[2];
  const float* feat = (const float*)d_in[3];
  const float* w1   = (const float*)d_in[4];
  const float* b1   = (const float*)d_in[5];
  const float* w2   = (const float*)d_in[6];
  const float* b2   = (const float*)d_in[7];
  const float* w3   = (const float*)d_in[8];
  const float* b3   = (const float*)d_in[9];
  const float* wmw  = (const float*)d_in[10];
  const float* bm   = (const float*)d_in[11];

  char* ws = (char*)d_ws;
  half_t* xbuf  = (half_t*)(ws + OFF_XBUF);
  half_t* a1    = (half_t*)(ws + OFF_A1);
  half_t* a2    = (half_t*)(ws + OFF_A2);
  half_t* a3    = (half_t*)(ws + OFF_A3);
  half_t* y1    = (half_t*)(ws + OFF_Y1);
  float*  guide = (float*)(ws + OFF_GUIDE);
  float*  meanI = (float*)(ws + OFF_MEANI);
  float*  invv  = (float*)(ws + OFF_INV);
  float*  P     = (float*)(ws + OFF_P);
  half_t* aF    = (half_t*)(ws + OFF_XBUF);   // reuse xbuf region
  half_t* bF    = (half_t*)(ws + OFF_BF);
  half_t* filt  = (half_t*)(ws + OFF_Y1);     // reuse y1 region

  float* out_mask = (float*)d_out;            // (1,1,256,256)
  float* out_feat = (float*)d_out + HW;       // (1,256,256,256)

  // 1. weight prep (covers a2's 1,638,400 elements -> 6400 blocks)
  k_prepw<<<dim3(6400), dim3(256), 0, stream>>>(w1, w2, w3, a1, a2, a3);
  // 2. upsample + concat + guide
  k_prep<<<dim3(HW / 256), dim3(256), 0, stream>>>(fg, bg, mask, feat, xbuf, guide);
  // 3. 1x1 conv #1 (WMMA), block covers M=256 x N=128
  k_gemm1<<<dim3(HW / 128), dim3(256), 0, stream>>>(a1, xbuf, b1, y1);
  // 4. 5x5 conv (implicit GEMM, WMMA), block covers M=256 x N=128
  k_gemm5<<<dim3(HW / 128), dim3(256), 0, stream>>>(a2, y1, b2, P);
  // 5. guided filter
  k_gfstats<<<dim3(HW / 256), dim3(256), 0, stream>>>(guide, meanI, invv);
  k_gfab<<<dim3(16, 16, 256), dim3(16, 16), 0, stream>>>(P, guide, meanI, invv, aF, bF);
  k_gfq<<<dim3(16, 16, 256), dim3(16, 16), 0, stream>>>(aF, bF, guide, filt);
  // 6. 1x1 conv #2 (WMMA) -> feat_2x output
  k_gemm3<<<dim3(HW / 128), dim3(256), 0, stream>>>(a3, filt, b3, out_feat);
  // 7. mask head
  k_mask<<<dim3(HW / 256), dim3(256), 0, stream>>>(out_feat, wmw, bm, out_mask);
}
